// Model_26663156973695
// MI455X (gfx1250) — matrix operand order repaired and checked by reading, not compiled, not run on hardware
//
#include <hip/hip_runtime.h>
#include <hip/hip_bf16.h>

typedef __attribute__((ext_vector_type(16))) _Float16 v16h;
typedef __attribute__((ext_vector_type(8)))  _Float16 v8h;
typedef __attribute__((ext_vector_type(8)))  float    v8f;

namespace {
constexpr int BATCH = 32;
constexpr int SEQ   = 197;
constexpr int SEQP  = 256;   // key length padded so K/32 is EVEN for all GEMMs
constexpr int HID   = 768;
constexpr int NHEAD = 12;
constexpr int HDIM  = 64;
constexpr int FFD   = 3072;
constexpr int NLAY  = 12;
constexpr int NPAT  = 196;
constexpr int NCLS  = 1000;
constexpr int MTOK  = BATCH * SEQ;   // 6304
}

// ---------------------------------------------------------------------------
// Generic batched GEMM:  C[M,N] = alpha * A[M,K] @ Bt[N,K]^T (+bias)(relu)(+resid)
// A row-major f16 (lda), Bt is B transposed, row-major f16 (ldb).
// Requires K % 64 == 0 (host guarantees: K in {64, 256, 768, 3072}).
//
// Blocking: 4 waves per block arranged 2x2; each wave owns a 64(M) x 32(N)
// register tile = 4x2 WMMA tiles (8 f32 accumulators). Block tile 128x64.
//
// OOB rows/cols handled by CLAMPING load row indices (an A row only feeds its
// own C row, a Bt row only its own C column; clamped duplicates never stored)
// -> branch-free K loop, EXEC all-ones at every WMMA.
//
// Fragments are PING-PONG double-buffered with an unroll-by-2 K loop: loads of
// step s+1 target the idle buffer while the 8 v_wmma of step s consume the
// live one -- no inter-buffer register copies, minimal hazard NOPs.
//
// Fragment layouts (A/B in contiguous k order, identical for both operands):
//   A (16x32 f16): lane l: m=l%16, hg=l/16; halves j = A[m][k0+hg*16+j]
//                                                        -> one 32B load
//   B (32x16 f16): lane l: n=l%16, hg=l/16; halves j = Bt[n][k0+hg*16+j]
//                                                        -> one 32B load
//   C/D (16x16 f32): reg r: m = hg*8+r, n = l%16
// ---------------------------------------------------------------------------
__global__ __launch_bounds__(128)
void gemm_wmma_f16(const _Float16* __restrict__ A, const _Float16* __restrict__ Bt,
                   const float* __restrict__ bias, const float* __restrict__ resid,
                   float* __restrict__ C, _Float16* __restrict__ C16,
                   int M, int N, int K, int lda, int ldb, int ldc,
                   long long sAb, long long sAh, long long sBb, long long sBh,
                   long long sCb, long long sCh, int nh, float alpha, int relu)
{
  const int z  = blockIdx.z;
  const int bb = z / nh;
  const int hh = z - bb * nh;
  A  += (long long)bb * sAb + (long long)hh * sAh;
  Bt += (long long)bb * sBb + (long long)hh * sBh;
  const long long coff = (long long)bb * sCb + (long long)hh * sCh;

  const int wave  = threadIdx.x >> 5;
  const int lane  = threadIdx.x & 31;
  const int waveM = wave >> 1;         // 0..1
  const int waveN = wave & 1;          // 0..1
  const int hg = lane >> 4;
  const int mr = lane & 15;

  const int tm0 = blockIdx.y * 8 + waveM * 4;  // first of 4 M-tiles (16 rows each)
  const int tn0 = blockIdx.x * 4 + waveN * 2;  // first of 2 N-tiles (16 cols each)

  const _Float16* arow[4];
#pragma unroll
  for (int i = 0; i < 4; ++i) {
    int m = tm0 * 16 + i * 16 + mr;
    if (m > M - 1) m = M - 1;          // clamp: result rows >= M never stored
    arow[i] = A + (long long)m * lda;
  }
  const _Float16* brow[2];
#pragma unroll
  for (int j = 0; j < 2; ++j) {
    int n = tn0 * 16 + j * 16 + mr;
    if (n > N - 1) n = N - 1;          // clamp: result cols >= N never stored
    brow[j] = Bt + (long long)n * ldb;
  }

  const int aoff = hg * 16;            // halves
  const int boff = hg * 16;

  v8f acc[4][2] = {};
  v16h a0b[4], b0b[2], a1b[4], b1b[2];

  auto loadA = [&](v16h* dst, int k0) {
#pragma unroll
    for (int i = 0; i < 4; ++i) {
      dst[i] = *(const v16h*)(arow[i] + k0 + aoff);
    }
  };
  auto loadB = [&](v16h* dst, int k0) {
#pragma unroll
    for (int j = 0; j < 2; ++j) dst[j] = *(const v16h*)(brow[j] + k0 + boff);
  };
  auto mma = [&](const v16h* a, const v16h* b) {
#pragma unroll
    for (int i = 0; i < 4; ++i)
#pragma unroll
      for (int j = 0; j < 2; ++j)
        acc[i][j] = __builtin_amdgcn_wmma_f32_16x16x32_f16(false, a[i], false, b[j],
                                                           (short)0, acc[i][j], false, false);
  };

  loadA(a0b, 0);
  loadB(b0b, 0);
  const int nsteps = K >> 5;           // even by construction
  for (int s = 0; s + 2 < nsteps; s += 2) {
    loadA(a1b, (s + 1) * 32);
    loadB(b1b, (s + 1) * 32);
    mma(a0b, b0b);
    loadA(a0b, (s + 2) * 32);
    loadB(b0b, (s + 2) * 32);
    mma(a1b, b1b);
  }
  loadA(a1b, (nsteps - 1) * 32);
  loadB(b1b, (nsteps - 1) * 32);
  mma(a0b, b0b);
  mma(a1b, b1b);

  // ---- epilogue ----
#pragma unroll
  for (int j = 0; j < 2; ++j) {
    const int nn = tn0 * 16 + j * 16 + mr;
    if (nn < N) {
      const float bv = bias ? bias[nn] : 0.0f;
#pragma unroll
      for (int i = 0; i < 4; ++i) {
#pragma unroll
        for (int r = 0; r < 8; ++r) {
          const int mm = tm0 * 16 + i * 16 + hg * 8 + r;
          if (mm < M) {
            float v = acc[i][j][r] * alpha + bv;
            if (relu) v = fmaxf(v, 0.0f);
            const long long o = coff + (long long)mm * ldc + nn;
            if (resid) v += resid[o];
            if (C)   C[o]   = v;
            if (C16) C16[o] = (_Float16)v;
          }
        }
      }
    }
  }
}

// ---------------------------------------------------------------------------
// LayerNorm over last dim (D=768); writes f32 and f16 copies.
// ---------------------------------------------------------------------------
__global__ __launch_bounds__(256)
void layernorm_f32(const float* __restrict__ in, const float* __restrict__ gamma,
                   const float* __restrict__ beta, float* __restrict__ out,
                   _Float16* __restrict__ out16, int D)
{
  __shared__ float red[256];
  const long long row = blockIdx.x;
  const float* x = in + row * D;
  const int t = threadIdx.x;

  float s = 0.0f;
  for (int i = t; i < D; i += 256) s += x[i];
  red[t] = s; __syncthreads();
  for (int o = 128; o > 0; o >>= 1) { if (t < o) red[t] += red[t + o]; __syncthreads(); }
  const float mean = red[0] / (float)D;
  __syncthreads();

  float v = 0.0f;
  for (int i = t; i < D; i += 256) { float d = x[i] - mean; v += d * d; }
  red[t] = v; __syncthreads();
  for (int o = 128; o > 0; o >>= 1) { if (t < o) red[t] += red[t + o]; __syncthreads(); }
  const float rstd = rsqrtf(red[0] / (float)D + 1e-5f);
  __syncthreads();

  for (int i = t; i < D; i += 256) {
    float y = (x[i] - mean) * rstd * gamma[i] + beta[i];
    out[row * D + i]   = y;
    out16[row * D + i] = (_Float16)y;
  }
}

// ---------------------------------------------------------------------------
// Softmax over 197 valid keys; writes f16 probs padded with zeros to 256.
// One block per (b, head, query).
// ---------------------------------------------------------------------------
__global__ __launch_bounds__(256)
void softmax_rows(const float* __restrict__ scores, _Float16* __restrict__ probs)
{
  __shared__ float red[256];
  const long long row = blockIdx.x;
  const float* x = scores + row * SEQP;
  _Float16* p = probs + row * SEQP;
  const int t = threadIdx.x;

  float m = -3.0e38f;
  for (int i = t; i < SEQ; i += 256) m = fmaxf(m, x[i]);
  red[t] = m; __syncthreads();
  for (int o = 128; o > 0; o >>= 1) { if (t < o) red[t] = fmaxf(red[t], red[t + o]); __syncthreads(); }
  m = red[0]; __syncthreads();

  float s = 0.0f;
  for (int i = t; i < SEQ; i += 256) s += __expf(x[i] - m);
  red[t] = s; __syncthreads();
  for (int o = 128; o > 0; o >>= 1) { if (t < o) red[t] += red[t + o]; __syncthreads(); }
  const float inv = 1.0f / red[0];

  for (int i = t; i < SEQP; i += 256)
    p[i] = (i < SEQ) ? (_Float16)(__expf(x[i] - m) * inv) : (_Float16)0.0f;
}

// ---------------------------------------------------------------------------
// Patch extraction matching the reference's linear reshape semantics.
// ---------------------------------------------------------------------------
__global__ __launch_bounds__(256)
void patchify_f16(const float* __restrict__ x, _Float16* __restrict__ out)
{
  const long long total = (long long)BATCH * NPAT * HID;
  long long idx = (long long)blockIdx.x * 256 + threadIdx.x;
  if (idx >= total) return;
  const int b = (int)(idx / ((long long)NPAT * HID));
  const int r = (int)(idx % ((long long)NPAT * HID));
  const int n = r / HID;
  const int d = r % HID;
  const int lin = n * HID + d;        // linear over (C, Hp, Wp, Py, Px)
  const int c  = lin / 50176;         // 196*256
  const int r1 = lin % 50176;
  const int hp = r1 / 3584;           // 14*256
  const int r2 = r1 % 3584;
  const int wp = r2 / 256;
  const int r3 = r2 % 256;
  const int py = r3 / 16;
  const int px = r3 % 16;
  out[idx] = (_Float16)x[(((long long)b * 3 + c) * 224 + (hp * 16 + py)) * 224 + (wp * 16 + px)];
}

__global__ __launch_bounds__(256)
void embed_add(const float* __restrict__ pe, const float* __restrict__ cls,
               const float* __restrict__ pos, float* __restrict__ h)
{
  const long long total = (long long)BATCH * SEQ * HID;
  long long idx = (long long)blockIdx.x * 256 + threadIdx.x;
  if (idx >= total) return;
  const int r = (int)(idx % ((long long)SEQ * HID));
  const int b = (int)(idx / ((long long)SEQ * HID));
  const int s = r / HID;
  const int j = r % HID;
  float v = (s == 0) ? cls[j] : pe[((long long)b * NPAT + (s - 1)) * HID + j];
  h[idx] = v + pos[s * HID + j];
}

// V[b,s,h*64+d] (f16) -> Vt[b,h,d,s_pad] (f16, zero padded to 256)
__global__ __launch_bounds__(256)
void v_transpose_f16(const _Float16* __restrict__ v, _Float16* __restrict__ vt)
{
  const long long total = (long long)BATCH * NHEAD * HDIM * SEQP;
  long long idx = (long long)blockIdx.x * 256 + threadIdx.x;
  if (idx >= total) return;
  const int s = (int)(idx % SEQP);
  long long q = idx / SEQP;
  const int d = (int)(q % HDIM);  q /= HDIM;
  const int h = (int)(q % NHEAD);
  const int b = (int)(q / NHEAD);
  _Float16 val = (_Float16)0.0f;
  if (s < SEQ) val = v[((long long)b * SEQ + s) * HID + h * HDIM + d];
  vt[idx] = val;
}

__global__ __launch_bounds__(256)
void gather_cls_f16(const float* __restrict__ h, _Float16* __restrict__ cls16)
{
  int idx = blockIdx.x * 256 + threadIdx.x;
  if (idx >= BATCH * HID) return;
  const int b = idx / HID;
  const int j = idx % HID;
  cls16[idx] = (_Float16)h[(long long)b * SEQ * HID + j];
}

// in f32 [K,N] (layer = blockIdx.z) -> out f16 [N,K]
__global__ __launch_bounds__(256)
void transpose_w_f16(const float* __restrict__ in, _Float16* __restrict__ out, int K, int N)
{
  const long long base = (long long)blockIdx.z * K * N;
  long long idx = (long long)blockIdx.x * 256 + threadIdx.x;
  if (idx >= (long long)K * N) return;
  const int k = (int)(idx / N);
  const int n = (int)(idx % N);
  out[base + (long long)n * K + k] = (_Float16)in[base + idx];
}

// ---------------------------------------------------------------------------
// Host side
// ---------------------------------------------------------------------------
static inline void gemm(hipStream_t st, const _Float16* A, const _Float16* Bt,
                        const float* bias, const float* resid, float* C, _Float16* C16,
                        int M, int N, int K, int lda, int ldb, int ldc,
                        long long sAb, long long sAh, long long sBb, long long sBh,
                        long long sCb, long long sCh, int batB, int batH,
                        float alpha, int relu)
{
  dim3 grid((unsigned)((N + 63) / 64), (unsigned)((M + 127) / 128), (unsigned)(batB * batH));
  gemm_wmma_f16<<<grid, 128, 0, st>>>(A, Bt, bias, resid, C, C16, M, N, K, lda, ldb, ldc,
                                      sAb, sAh, sBb, sBh, sCb, sCh, batH, alpha, relu);
}

extern "C" void kernel_launch(void* const* d_in, const int* in_sizes, int n_in,
                              void* d_out, int out_size, void* d_ws, size_t ws_size,
                              hipStream_t stream)
{
  (void)in_sizes; (void)n_in; (void)out_size; (void)ws_size;
  const float* x       = (const float*)d_in[0];
  const float* w_patch = (const float*)d_in[1];
  const float* b_patch = (const float*)d_in[2];
  const float* cls_tok = (const float*)d_in[3];
  const float* pos_emb = (const float*)d_in[4];
  const float* ln1_s   = (const float*)d_in[5];
  const float* ln1_b   = (const float*)d_in[6];
  const float* wq = (const float*)d_in[7];
  const float* bq = (const float*)d_in[8];
  const float* wk = (const float*)d_in[9];
  const float* bk = (const float*)d_in[10];
  const float* wv = (const float*)d_in[11];
  const float* bv = (const float*)d_in[12];
  const float* wo = (const float*)d_in[13];
  const float* bo = (const float*)d_in[14];
  const float* ln2_s = (const float*)d_in[15];
  const float* ln2_b = (const float*)d_in[16];
  const float* w1 = (const float*)d_in[17];
  const float* b1 = (const float*)d_in[18];
  const float* w2 = (const float*)d_in[19];
  const float* b2 = (const float*)d_in[20];
  const float* w_fc = (const float*)d_in[21];
  const float* b_fc = (const float*)d_in[22];
  float* out = (float*)d_out;

  size_t off = 0;
  auto alloc = [&](size_t bytes) -> void* {
    off = (off + 255) & ~(size_t)255;
    void* p = (char*)d_ws + off;
    off += bytes;
    return p;
  };

  // Persistent f16 transposed weights (rebuilt every launch: deterministic)
  _Float16* wpatch_t = (_Float16*)alloc((size_t)HID * HID * 2);
  _Float16* wq_t = (_Float16*)alloc((size_t)NLAY * HID * HID * 2);
  _Float16* wk_t = (_Float16*)alloc((size_t)NLAY * HID * HID * 2);
  _Float16* wv_t = (_Float16*)alloc((size_t)NLAY * HID * HID * 2);
  _Float16* wo_t = (_Float16*)alloc((size_t)NLAY * HID * HID * 2);
  _Float16* w1_t = (_Float16*)alloc((size_t)NLAY * HID * FFD * 2);
  _Float16* w2_t = (_Float16*)alloc((size_t)NLAY * FFD * HID * 2);
  _Float16* wfc_t = (_Float16*)alloc((size_t)NCLS * HID * 2);

  // Activations
  float*    h        = (float*)alloc((size_t)MTOK * HID * 4);
  float*    hln      = (float*)alloc((size_t)MTOK * HID * 4);
  _Float16* hln16    = (_Float16*)alloc((size_t)MTOK * HID * 2);
  _Float16* q16      = (_Float16*)alloc((size_t)MTOK * HID * 2);
  _Float16* k16      = (_Float16*)alloc((size_t)MTOK * HID * 2);
  _Float16* v16buf   = (_Float16*)alloc((size_t)MTOK * HID * 2);
  _Float16* ctx16    = (_Float16*)alloc((size_t)MTOK * HID * 2);
  _Float16* vT16     = (_Float16*)alloc((size_t)BATCH * NHEAD * HDIM * SEQP * 2);
  float*    scores   = (float*)alloc((size_t)BATCH * NHEAD * SEQ * SEQP * 4);
  _Float16* probs16  = (_Float16*)alloc((size_t)BATCH * NHEAD * SEQ * SEQP * 2);
  _Float16* ff16     = (_Float16*)alloc((size_t)MTOK * FFD * 2);
  _Float16* patches16= (_Float16*)alloc((size_t)BATCH * NPAT * HID * 2);
  float*    ptmp     = (float*)alloc((size_t)BATCH * NPAT * HID * 4);
  _Float16* cls16    = (_Float16*)alloc((size_t)BATCH * HID * 2);

  // --- weight conversion / transposition (f32 [K,N] -> f16 [N,K]) ---
  auto T = [&](const float* in, _Float16* o, int K, int N, int layers) {
    dim3 g((unsigned)(((long long)K * N + 255) / 256), 1u, (unsigned)layers);
    transpose_w_f16<<<g, 256, 0, stream>>>(in, o, K, N);
  };
  T(w_patch, wpatch_t, HID, HID, 1);
  T(wq, wq_t, HID, HID, NLAY);
  T(wk, wk_t, HID, HID, NLAY);
  T(wv, wv_t, HID, HID, NLAY);
  T(wo, wo_t, HID, HID, NLAY);
  T(w1, w1_t, HID, FFD, NLAY);
  T(w2, w2_t, FFD, HID, NLAY);
  T(w_fc, wfc_t, HID, NCLS, 1);

  // --- patch embedding ---
  {
    long long n = (long long)BATCH * NPAT * HID;
    patchify_f16<<<(unsigned)((n + 255) / 256), 256, 0, stream>>>(x, patches16);
    gemm(stream, patches16, wpatch_t, b_patch, nullptr, ptmp, nullptr,
         BATCH * NPAT, HID, HID, HID, HID, HID, 0,0,0,0,0,0, 1,1, 1.0f, 0);
    long long m = (long long)BATCH * SEQ * HID;
    embed_add<<<(unsigned)((m + 255) / 256), 256, 0, stream>>>(ptmp, cls_tok, pos_emb, h);
  }

  // --- transformer layers ---
  for (int l = 0; l < NLAY; ++l) {
    const _Float16* Wq = wq_t + (size_t)l * HID * HID;
    const _Float16* Wk = wk_t + (size_t)l * HID * HID;
    const _Float16* Wv = wv_t + (size_t)l * HID * HID;
    const _Float16* Wo = wo_t + (size_t)l * HID * HID;
    const _Float16* W1 = w1_t + (size_t)l * HID * FFD;
    const _Float16* W2 = w2_t + (size_t)l * FFD * HID;

    // LN1 (reference: h = LN(h); residual taken from post-LN value)
    layernorm_f32<<<MTOK, 256, 0, stream>>>(h, ln1_s + l * HID, ln1_b + l * HID,
                                            hln, hln16, HID);
    // Q, K, V projections (f16 outputs feed attention WMMAs)
    gemm(stream, hln16, Wq, bq + l * HID, nullptr, nullptr, q16,
         MTOK, HID, HID, HID, HID, HID, 0,0,0,0,0,0, 1,1, 1.0f, 0);
    gemm(stream, hln16, Wk, bk + l * HID, nullptr, nullptr, k16,
         MTOK, HID, HID, HID, HID, HID, 0,0,0,0,0,0, 1,1, 1.0f, 0);
    gemm(stream, hln16, Wv, bv + l * HID, nullptr, nullptr, v16buf,
         MTOK, HID, HID, HID, HID, HID, 0,0,0,0,0,0, 1,1, 1.0f, 0);

    // scores[b,h,q,k] = Q·K^T / 8   (batched over 32*12, K-dim = 64)
    gemm(stream, q16, k16, nullptr, nullptr, scores, nullptr,
         SEQ, SEQ, HDIM, HID, HID, SEQP,
         (long long)SEQ * HID, HDIM, (long long)SEQ * HID, HDIM,
         (long long)NHEAD * SEQ * SEQP, (long long)SEQ * SEQP,
         BATCH, NHEAD, 0.125f, 0);

    softmax_rows<<<BATCH * NHEAD * SEQ, 256, 0, stream>>>(scores, probs16);

    {
      long long n = (long long)BATCH * NHEAD * HDIM * SEQP;
      v_transpose_f16<<<(unsigned)((n + 255) / 256), 256, 0, stream>>>(v16buf, vT16);
    }

    // ctx[b,s,h*64+d] = probs @ V   (K-dim = 256 padded; zeros contribute 0)
    gemm(stream, probs16, vT16, nullptr, nullptr, nullptr, ctx16,
         SEQ, HDIM, SEQP, SEQP, SEQP, HID,
         (long long)NHEAD * SEQ * SEQP, (long long)SEQ * SEQP,
         (long long)NHEAD * HDIM * SEQP, (long long)HDIM * SEQP,
         (long long)SEQ * HID, HDIM,
         BATCH, NHEAD, 1.0f, 0);

    // out projection + residual: h = hln + ctx @ Wo + bo
    gemm(stream, ctx16, Wo, bo + l * HID, hln, h, nullptr,
         MTOK, HID, HID, HID, HID, HID, 0,0,0,0,0,0, 1,1, 1.0f, 0);

    // LN2
    layernorm_f32<<<MTOK, 256, 0, stream>>>(h, ln2_s + l * HID, ln2_b + l * HID,
                                            hln, hln16, HID);
    // FF1: relu(hln @ W1 + b1) -> f16
    gemm(stream, hln16, W1, b1 + l * FFD, nullptr, nullptr, ff16,
         MTOK, FFD, HID, HID, HID, FFD, 0,0,0,0,0,0, 1,1, 1.0f, 1);
    // FF2 + residual: h = hln + ff @ W2 + b2
    gemm(stream, ff16, W2, b2 + l * HID, hln, h, nullptr,
         MTOK, HID, FFD, FFD, FFD, HID, 0,0,0,0,0,0, 1,1, 1.0f, 0);
  }

  // --- classifier head on CLS token ---
  gather_cls_f16<<<(BATCH * HID + 255) / 256, 256, 0, stream>>>(h, cls16);
  gemm(stream, cls16, wfc_t, b_fc, nullptr, out, nullptr,
       BATCH, NCLS, HID, HID, HID, NCLS, 0,0,0,0,0,0, 1,1, 1.0f, 0);
}
